// Qwen3MoeSparseMoeBlock_9225589752214
// MI455X (gfx1250) — matrix operand order repaired and checked by reading, not compiled, not run on hardware
//
#include <hip/hip_runtime.h>
#include <hip/hip_bf16.h>

// Problem constants (match reference)
#define HD   2048   // hidden
#define NE   64     // experts
#define II   768    // moe intermediate
#define I2   1536   // 2*I
#define MT   512    // tokens (2*256)
#define TOPK 8
#define TILE_M 16
#define KC   256    // k-chunk staged in LDS

typedef __bf16    v16bf __attribute__((ext_vector_type(16)));
typedef float     v8f   __attribute__((ext_vector_type(8)));
typedef float     f32x4 __attribute__((ext_vector_type(4)));
typedef unsigned  u32x4 __attribute__((ext_vector_type(4)));

union ABu {
    v16bf  v;
    __bf16 h[16];
    u32x4  q[2];
};

// ---------------------------------------------------------------------------
// Router: logits -> softmax -> top-8 -> renormalize -> per-expert lists
// One 64-thread block per token. gate_w is only 512KB -> lives in L2.
// ---------------------------------------------------------------------------
__global__ __launch_bounds__(64) void moe_router_kernel(
    const float* __restrict__ x, const float* __restrict__ gate_w,
    int* __restrict__ expert_count, int* __restrict__ expert_tokens,
    float* __restrict__ expert_wts)
{
    const int m = blockIdx.x;
    __shared__ float sx[HD];
    __shared__ float sv[NE];

    for (int h = threadIdx.x; h < HD; h += 64)
        sx[h] = x[(size_t)m * HD + h];
    __syncthreads();

    // each thread = one expert logit
    {
        const float* w = gate_w + (size_t)threadIdx.x * HD;
        float acc = 0.f;
        #pragma unroll 8
        for (int h = 0; h < HD; ++h) acc = fmaf(sx[h], w[h], acc);
        sv[threadIdx.x] = acc;
    }
    __syncthreads();

    if (threadIdx.x == 0) {
        float mx = -1e30f;
        for (int i = 0; i < NE; ++i) mx = fmaxf(mx, sv[i]);
        for (int i = 0; i < NE; ++i) sv[i] = __expf(sv[i] - mx);
        // top-8 by repeated argmax; softmax denominator cancels under renorm
        int   ids[TOPK];
        float ws [TOPK];
        float wsum = 0.f;
        for (int k = 0; k < TOPK; ++k) {
            int bi = 0; float bv = -1.f;
            for (int i = 0; i < NE; ++i) {
                float v = sv[i];
                if (v > bv) { bv = v; bi = i; }
            }
            ids[k] = bi; ws[k] = bv; wsum += bv; sv[bi] = -1.f;
        }
        const float inv = 1.f / wsum;
        for (int k = 0; k < TOPK; ++k) {
            int pos = atomicAdd(expert_count + ids[k], 1);
            expert_tokens[ids[k] * MT + pos] = m;
            expert_wts  [ids[k] * MT + pos] = ws[k] * inv;
        }
    }
}

// ---------------------------------------------------------------------------
// Fused expert FFN: x-tile -> (gate|up) -> silu*up -> down -> weighted combine
// Block = 256 threads (8 waves). blockIdx.x = expert, blockIdx.y = token tile.
// A (activations) staged bf16 in LDS, B (weights) streamed fp32 -> bf16 regs.
// All matmuls via v_wmma_f32_16x16x32_bf16, fp32 accumulation.
// ---------------------------------------------------------------------------
__global__ __launch_bounds__(256) void moe_ffn_kernel(
    const float* __restrict__ x,   const float* __restrict__ w13,
    const float* __restrict__ w2,  const int* __restrict__ expert_count,
    const int* __restrict__ expert_tokens, const float* __restrict__ expert_wts,
    float* __restrict__ out)
{
    const int e    = blockIdx.x;
    const int cnt  = expert_count[e];
    const int row0 = blockIdx.y * TILE_M;
    if (row0 >= cnt) return;
    const int nrows = (cnt - row0 < TILE_M) ? (cnt - row0) : TILE_M;

    const int lane = threadIdx.x & 31;
    const int wave = threadIdx.x >> 5;          // 0..7
    const int arow  = lane & 15;                // A row / C column lane
    const int klo_a = (lane < 16) ? 0 : 16;     // A-layout K half offset
    const int klo_b = (lane < 16) ? 0 : 16;     // B-layout K half offset
    const int mhi   = (lane < 16) ? 0 : 8;      // C/D M offset per lane half

    __shared__ __bf16 sX  [TILE_M][KC + 8];     // x-tile k-chunk (bf16)
    __shared__ __bf16 sAct[TILE_M][II + 8];     // silu(g)*u activations (bf16)
    __shared__ int    sTok[TILE_M];
    __shared__ float  sWt [TILE_M];

    if (threadIdx.x < TILE_M) {
        int r = threadIdx.x;
        if (r < nrows) {
            sTok[r] = expert_tokens[e * MT + row0 + r];
            sWt [r] = expert_wts  [e * MT + row0 + r];
        } else {
            sTok[r] = 0; sWt[r] = 0.f;
        }
    }

    // ---------------- Phase 1: h = x @ w13[e]^T , act = silu(g)*u -----------
    v8f accG[6] = {}, accU[6] = {};
    const int nbase = wave * 96;                // this wave's 96 g/u columns

    for (int kc = 0; kc < HD; kc += KC) {
        __syncthreads();
        // stage x k-chunk as bf16 (rows past nrows zeroed)
        for (int idx = threadIdx.x; idx < TILE_M * KC; idx += 256) {
            int r = idx >> 8, c = idx & (KC - 1);
            float v = (r < nrows) ? x[(size_t)sTok[r] * HD + kc + c] : 0.f;
            sX[r][c] = (__bf16)v;
        }
        __syncthreads();

        for (int kk = 0; kk < KC; kk += 32) {
            ABu a;
            const u32x4* xr = (const u32x4*)(&sX[arow][0]);
            a.q[0] = xr[(kk +      klo_a) >> 3];
            a.q[1] = xr[(kk +  8 + klo_a) >> 3];

            const int kglob = kc + kk + klo_b;
            #pragma unroll
            for (int t = 0; t < 6; ++t) {
                const int n = nbase + t * 16 + (lane & 15);
                const float* wg = w13 + ((size_t)e * I2 + n)       * HD + kglob;
                const float* wu = w13 + ((size_t)e * I2 + II + n)  * HD + kglob;
                ABu bg, bu;
                #pragma unroll
                for (int j = 0; j < 4; ++j) {
                    f32x4 qg = *(const f32x4*)(wg + 4 * j);
                    f32x4 qu = *(const f32x4*)(wu + 4 * j);
                    bg.h[4*j+0] = (__bf16)qg.x; bg.h[4*j+1] = (__bf16)qg.y;
                    bg.h[4*j+2] = (__bf16)qg.z; bg.h[4*j+3] = (__bf16)qg.w;
                    bu.h[4*j+0] = (__bf16)qu.x; bu.h[4*j+1] = (__bf16)qu.y;
                    bu.h[4*j+2] = (__bf16)qu.z; bu.h[4*j+3] = (__bf16)qu.w;
                }
                accG[t] = __builtin_amdgcn_wmma_f32_16x16x32_bf16(
                    false, a.v, false, bg.v, (short)0, accG[t], false, false);
                accU[t] = __builtin_amdgcn_wmma_f32_16x16x32_bf16(
                    false, a.v, false, bu.v, (short)0, accU[t], false, false);
            }
        }
    }

    // silu(g) * u -> LDS (bf16)
    #pragma unroll
    for (int t = 0; t < 6; ++t) {
        #pragma unroll
        for (int r = 0; r < 8; ++r) {
            float g = accG[t][r];
            float u = accU[t][r];
            float s = g / (1.0f + __expf(-g));
            sAct[r + mhi][nbase + t * 16 + (lane & 15)] = (__bf16)(s * u);
        }
    }
    __syncthreads();

    // ---------------- Phase 2: out += wt * (act @ w2[e]^T) ------------------
    for (int half = 0; half < 2; ++half) {
        v8f acc[8] = {};
        const int ncol0 = wave * 256 + half * 128;   // this wave's 128 H cols

        for (int i0 = 0; i0 < II; i0 += 32) {
            ABu a;
            const u32x4* ar = (const u32x4*)(&sAct[arow][0]);
            a.q[0] = ar[(i0 +      klo_a) >> 3];
            a.q[1] = ar[(i0 +  8 + klo_a) >> 3];

            const int ig = i0 + klo_b;
            #pragma unroll
            for (int t = 0; t < 8; ++t) {
                const int n = ncol0 + t * 16 + (lane & 15);
                const float* wp = w2 + ((size_t)e * HD + n) * II + ig;
                ABu b;
                #pragma unroll
                for (int j = 0; j < 4; ++j) {
                    f32x4 q = *(const f32x4*)(wp + 4 * j);
                    b.h[4*j+0] = (__bf16)q.x; b.h[4*j+1] = (__bf16)q.y;
                    b.h[4*j+2] = (__bf16)q.z; b.h[4*j+3] = (__bf16)q.w;
                }
                acc[t] = __builtin_amdgcn_wmma_f32_16x16x32_bf16(
                    false, a.v, false, b.v, (short)0, acc[t], false, false);
            }
        }

        // weighted combine into out (fire-and-forget f32 atomics)
        #pragma unroll
        for (int t = 0; t < 8; ++t) {
            const int n = ncol0 + t * 16 + (lane & 15);
            #pragma unroll
            for (int r = 0; r < 8; ++r) {
                const int m = r + mhi;
                if (m < nrows) {
                    atomicAdd(&out[(size_t)sTok[m] * HD + n], acc[t][r] * sWt[m]);
                }
            }
        }
    }
}

// ---------------------------------------------------------------------------
extern "C" void kernel_launch(void* const* d_in, const int* in_sizes, int n_in,
                              void* d_out, int out_size, void* d_ws, size_t ws_size,
                              hipStream_t stream) {
    const float* x      = (const float*)d_in[0];   // [2,256,2048]
    const float* gate_w = (const float*)d_in[1];   // [64,2048]
    const float* w13    = (const float*)d_in[2];   // [64,1536,2048]
    const float* w2     = (const float*)d_in[3];   // [64,2048,768]
    float* out = (float*)d_out;

    // workspace: counts (64 i32) | token lists (64*512 i32) | weights (64*512 f32)
    int*   expert_count  = (int*)d_ws;
    int*   expert_tokens = expert_count + NE;
    float* expert_wts    = (float*)(expert_tokens + NE * MT);

    hipMemsetAsync(expert_count, 0, NE * sizeof(int), stream);
    hipMemsetAsync(out, 0, (size_t)out_size * sizeof(float), stream);

    moe_router_kernel<<<MT, 64, 0, stream>>>(x, gate_w, expert_count,
                                             expert_tokens, expert_wts);

    dim3 grid(NE, MT / TILE_M);  // experts x max token-tiles; empty tiles exit
    moe_ffn_kernel<<<grid, 256, 0, stream>>>(x, w13, w2, expert_count,
                                             expert_tokens, expert_wts, out);
}
